// HTMAttention_13022340841898
// MI455X (gfx1250) — matrix operand order repaired and checked by reading, not compiled, not run on hardware
//
#include <hip/hip_runtime.h>
#include <math.h>

// ---------------- problem constants ----------------
#define B_     2
#define QL     128
#define D_     512
#define MLEN   8192
#define NC     256      // MLEN / CHUNK
#define CHK    32
#define TOPK_  8
#define H_     8
#define DH_    64
#define INNER  512      // H_*DH_
#define KV_N   1024     // 2*INNER

typedef __attribute__((ext_vector_type(16))) __bf16 v16bf;
typedef __attribute__((ext_vector_type(8)))  __bf16 v8bf;
typedef __attribute__((ext_vector_type(8)))  float  v8f;

#define CAT16(lo, hi) __builtin_shufflevector(lo, hi, 0, 1, 2, 3, 4, 5, 6, 7, \
                                              8, 9, 10, 11, 12, 13, 14, 15)

// ---------------- chunk summaries: summ[b,j,d] = mean_c mem[b,j,c,d] -------
__global__ __launch_bounds__(256) void summ_kernel(const float* __restrict__ mem,
                                                   float* __restrict__ summ) {
    int gid = blockIdx.x * 256 + threadIdx.x;          // over B_*NC*D_
    int d = gid % D_;
    int j = (gid / D_) % NC;
    int b = gid / (D_ * NC);
    const float* p = mem + ((size_t)(b * MLEN + j * CHK)) * D_ + d;
    float acc = 0.f;
    #pragma unroll 8
    for (int c = 0; c < CHK; ++c) acc += p[(size_t)c * D_];
    summ[gid] = acc * (1.0f / CHK);
}

// ---------------- scalar projection: out[r,n] = (sum_k X[r,k]W[k,n] + b[n])*s
__global__ __launch_bounds__(256) void linproj_kernel(const float* __restrict__ X,
                                                      const float* __restrict__ W,
                                                      const float* __restrict__ bias,
                                                      float* __restrict__ out,
                                                      int N, int Kd, float scale) {
    int gid = blockIdx.x * 256 + threadIdx.x;
    int col = gid % N;
    int row = gid / N;
    const float* x = X + (size_t)row * Kd;
    const float* w = W + col;
    float acc = 0.f;
    for (int k = 0; k < Kd; ++k) acc += x[k] * w[(size_t)k * N];
    if (bias) acc += bias[col];
    out[gid] = acc * scale;
}

// ------- per-query: sim over 256 chunks, top-8, softmax weights -----------
__global__ __launch_bounds__(256) void topk_kernel(const float* __restrict__ sq,
                                                   const float* __restrict__ sk,
                                                   int* __restrict__ tidx,
                                                   float* __restrict__ tw) {
    __shared__ float qrow[D_];
    __shared__ float simv[NC];
    __shared__ float rv[NC];
    __shared__ int   ri[NC];
    __shared__ int   bidx;
    const int tid = threadIdx.x;
    const int bi  = blockIdx.x;                 // over B_*QL
    const int b   = bi / QL;
    for (int t = tid; t < D_; t += 256) qrow[t] = sq[(size_t)bi * D_ + t];
    __syncthreads();
    const float* skr = sk + ((size_t)(b * NC + tid)) * D_;
    float acc = 0.f;
    for (int k = 0; k < D_; ++k) acc += qrow[k] * skr[k];
    simv[tid] = acc * 0.044194173824159216f;    // 512^-0.5
    __syncthreads();
    float logits[TOPK_]; int ids[TOPK_];
    for (int kk = 0; kk < TOPK_; ++kk) {
        rv[tid] = simv[tid]; ri[tid] = tid;
        __syncthreads();
        for (int s = 128; s > 0; s >>= 1) {
            if (tid < s) {
                bool take = (rv[tid + s] > rv[tid]) ||
                            (rv[tid + s] == rv[tid] && ri[tid + s] < ri[tid]);
                if (take) { rv[tid] = rv[tid + s]; ri[tid] = ri[tid + s]; }
            }
            __syncthreads();
        }
        if (tid == 0) { bidx = ri[0]; logits[kk] = rv[0]; ids[kk] = ri[0]; }
        __syncthreads();
        if (tid == bidx) simv[tid] = -3.4e38f;
        __syncthreads();
    }
    if (tid == 0) {
        float m = logits[0];
        for (int k = 1; k < TOPK_; ++k) m = fmaxf(m, logits[k]);
        float s = 0.f, w[TOPK_];
        for (int k = 0; k < TOPK_; ++k) { w[k] = __expf(logits[k] - m); s += w[k]; }
        float inv = 1.0f / s;
        for (int k = 0; k < TOPK_; ++k) {
            tidx[bi * TOPK_ + k] = ids[k];
            tw[bi * TOPK_ + k]   = w[k] * inv;
        }
    }
}

// -------- abf[b,m,d] = bf16(mem + sinusoidal_pos(c=m%32, d)) ---------------
__global__ __launch_bounds__(256) void prep_abf_kernel(const float* __restrict__ mem,
                                                       __bf16* __restrict__ abf) {
    int gid = blockIdx.x * 256 + threadIdx.x;     // over B_*MLEN*D_
    int d = gid % D_;
    int m = (gid / D_) % MLEN;
    int c = m & (CHK - 1);
    float seq = (float)(CHK - 1 - c);
    int f = d & 255;                               // d<256: sin, else cos (same freq idx)
    float invf = __expf(-(float)f * (9.210340371976184f / 256.0f)); // 10000^(-f/256)
    float x = seq * invf;
    float pos = (d < 256) ? __sinf(x) : __cosf(x);
    abf[gid] = (__bf16)(mem[gid] + pos);
}

// -------- transpose+convert: WT[n*K+k] = bf16(W[k*N+n]) --------------------
__global__ __launch_bounds__(256) void transposeW_kernel(const float* __restrict__ W,
                                                         __bf16* __restrict__ WT,
                                                         int Kd, int N) {
    int gid = blockIdx.x * 256 + threadIdx.x;     // over Kd*N
    int k = gid / N;
    int n = gid % N;
    WT[(size_t)n * Kd + k] = (__bf16)W[(size_t)k * N + n];
}

// -------- WMMA GEMM fragment helpers ---------------------------------------
struct Frags {
    v16bf a[4];
    v16bf b;
};

__device__ __forceinline__ void load_frags(Frags& f, const __bf16* brow,
                                           const __bf16* const* a0,
                                           const __bf16* const* a1, int kb) {
    f.b = *(const v16bf*)(brow + kb);
    #pragma unroll
    for (int g = 0; g < 4; ++g) {
        v8bf lo = *(const v8bf*)(a0[g] + kb);
        v8bf hi = *(const v8bf*)(a1[g] + kb);
        f.a[g] = CAT16(lo, hi);
    }
}

__device__ __forceinline__ void do_wmma(v8f* acc, const Frags& f) {
    #pragma unroll
    for (int g = 0; g < 4; ++g)
        acc[g] = __builtin_amdgcn_wmma_f32_16x16x32_bf16(
                    false, f.a[g], false, f.b, (short)0, acc[g], false, false);
}

// -------- WMMA GEMM: C[M,N] = A[M,Kd] bf16 (row-major) * B, with BT[N,Kd] --
// One wave computes a 64x16 strip: 4 M-register-blocked accumulators share
// each B fragment. Ping-pong double buffering (manually unrolled x2) keeps
// next K-step's loads in flight during the current WMMAs with no buffer-copy
// movs in the steady state.
template <typename OUT_T>
__global__ __launch_bounds__(256) void wmma_gemm_kernel(const __bf16* __restrict__ A,
                                                        const __bf16* __restrict__ BT,
                                                        OUT_T* __restrict__ C,
                                                        int M, int N, int Kd) {
    const int wave = threadIdx.x >> 5;
    const int lane = threadIdx.x & 31;
    const int l15  = lane & 15;
    const int kh   = lane >> 4;                    // 0 or 1 (K-half select)
    const int Mbase = blockIdx.y * 64;
    const int Nbase = blockIdx.x * 128 + wave * 16;

    // B fragment: col = Nbase+l15; contiguous 16 K-values at kb + kh*16
    const __bf16* brow = BT + (size_t)(Nbase + l15) * Kd + kh * 16;
    // A fragments: rows Mbase + 16*g + l15; per lane two 8-elem runs
    const __bf16* a0[4];
    const __bf16* a1[4];
    #pragma unroll
    for (int g = 0; g < 4; ++g) {
        const __bf16* arow = A + (size_t)(Mbase + 16 * g + l15) * Kd;
        a0[g] = arow + kh * 16;
        a1[g] = arow + kh * 16 + 8;
    }

    v8f acc[4] = {};
    Frags f0, f1;
    load_frags(f0, brow, a0, a1, 0);               // K-step 0
    int kb = 32;
    // steady state: invariant at loop entry: f0 holds K-step (kb-32)
    for (; kb + 32 < Kd; kb += 64) {
        __builtin_prefetch(brow + kb + 224, 0, 3); // global_prefetch_b8 (near)
        load_frags(f1, brow, a0, a1, kb);          // issue loads for step kb
        do_wmma(acc, f0);                          // consume step kb-32
        load_frags(f0, brow, a0, a1, kb + 32);     // issue loads for step kb+32
        do_wmma(acc, f1);                          // consume step kb
    }
    if (kb < Kd) {                                 // one unconsumed + one unloaded
        load_frags(f1, brow, a0, a1, kb);
        do_wmma(acc, f0);
        do_wmma(acc, f1);
    } else {                                       // single tail step in f0
        do_wmma(acc, f0);
    }

    // D layout: col = l15; VGPR r -> row base + r + 8*kh
    const int col = Nbase + l15;
    #pragma unroll
    for (int g = 0; g < 4; ++g) {
        const int rbase = Mbase + 16 * g + 8 * kh;
        #pragma unroll
        for (int r = 0; r < 8; ++r)
            C[(size_t)(rbase + r) * N + col] = (OUT_T)acc[g][r];
    }
}

// -------- inner attention: block = (b,i,slot), wave = head -----------------
// kv_all is bf16 [b, chunk, c, 1024]; math in f32.
__global__ __launch_bounds__(256) void attn_kernel(const float* __restrict__ qproj,
                                                   const __bf16* __restrict__ kv_all,
                                                   const int* __restrict__ tidx,
                                                   __bf16* __restrict__ obf) {
    const int gid  = blockIdx.x;                  // over B_*QL*TOPK_
    const int k    = gid % TOPK_;
    const int i    = (gid / TOPK_) % QL;
    const int b    = gid / (TOPK_ * QL);
    const int h    = threadIdx.x >> 5;
    const int lane = threadIdx.x & 31;            // = key position c

    const int j = tidx[(b * QL + i) * TOPK_ + k];
    const float* qr = qproj + ((size_t)(b * QL + i)) * INNER + h * DH_;   // pre-scaled
    const __bf16* kvb = kv_all + ((size_t)(b * NC + j) * CHK) * KV_N;

    const __bf16* Krow = kvb + (size_t)lane * KV_N + h * DH_;
    float s = 0.f;
    #pragma unroll
    for (int dd0 = 0; dd0 < DH_; dd0 += 8) {
        v8bf kk8 = *(const v8bf*)(Krow + dd0);
        #pragma unroll
        for (int t = 0; t < 8; ++t) s += qr[dd0 + t] * (float)kk8[t];
    }

    float m = s;
    for (int off = 16; off > 0; off >>= 1) m = fmaxf(m, __shfl_xor(m, off, 32));
    float p = __expf(s - m);
    float sum = p;
    for (int off = 16; off > 0; off >>= 1) sum += __shfl_xor(sum, off, 32);
    p /= sum;

    float o0 = 0.f, o1 = 0.f;
    for (int c = 0; c < CHK; ++c) {
        float pc = __shfl(p, c, 32);
        const __bf16* Vrow = kvb + (size_t)c * KV_N + INNER + h * DH_;
        o0 += pc * (float)Vrow[lane];
        o1 += pc * (float)Vrow[lane + 32];
    }
    const size_t orow = ((size_t)(b * TOPK_ + k) * QL + i);
    obf[orow * INNER + h * DH_ + lane]      = (__bf16)o0;
    obf[orow * INNER + h * DH_ + lane + 32] = (__bf16)o1;
}

// -------- combine: out[b,i,d] = sum_k w[b,i,k]*(oproj[b,k,i,d] + bo[d]) ----
__global__ __launch_bounds__(256) void combine_kernel(const float* __restrict__ oproj,
                                                      const float* __restrict__ tw,
                                                      const float* __restrict__ bo,
                                                      float* __restrict__ out) {
    int gid = blockIdx.x * 256 + threadIdx.x;     // over B_*QL*D_
    int d = gid % D_;
    int i = (gid / D_) % QL;
    int b = gid / (D_ * QL);
    float bias = bo[d];
    float acc = 0.f;
    #pragma unroll
    for (int k = 0; k < TOPK_; ++k) {
        float w = tw[(b * QL + i) * TOPK_ + k];
        size_t row = ((size_t)(b * TOPK_ + k) * QL + i);
        acc += w * (oproj[row * D_ + d] + bias);
    }
    out[gid] = acc;
}

// ---------------------------------------------------------------------------
extern "C" void kernel_launch(void* const* d_in, const int* in_sizes, int n_in,
                              void* d_out, int out_size, void* d_ws, size_t ws_size,
                              hipStream_t stream) {
    const float* queries  = (const float*)d_in[0];
    const float* memories = (const float*)d_in[1];
    const float* Wsq = (const float*)d_in[2];
    const float* bsq = (const float*)d_in[3];
    const float* Wsk = (const float*)d_in[4];
    const float* bsk = (const float*)d_in[5];
    const float* Wq  = (const float*)d_in[6];
    const float* Wkv = (const float*)d_in[7];
    const float* Wo  = (const float*)d_in[8];
    const float* bo  = (const float*)d_in[9];
    float* out = (float*)d_out;

    // workspace carve-up (aligned to 256B)
    char* ws = (char*)d_ws;
    size_t off = 0;
    auto take = [&](size_t bytes) -> void* {
        void* p = ws + off;
        off = (off + bytes + 255) & ~(size_t)255;
        return p;
    };
    float*  summ   = (float*) take((size_t)B_ * NC * D_ * 4);          //  1 MB
    float*  sq     = (float*) take((size_t)B_ * QL * D_ * 4);          // .5 MB
    float*  sk     = (float*) take((size_t)B_ * NC * D_ * 4);          //  1 MB
    float*  qproj  = (float*) take((size_t)B_ * QL * INNER * 4);       // .5 MB
    int*    tidx   = (int*)   take((size_t)B_ * QL * TOPK_ * 4);
    float*  tw     = (float*) take((size_t)B_ * QL * TOPK_ * 4);
    __bf16* abf    = (__bf16*)take((size_t)B_ * MLEN * D_ * 2);        // 17 MB
    __bf16* wkvT   = (__bf16*)take((size_t)D_ * KV_N * 2);             //  1 MB
    __bf16* woT    = (__bf16*)take((size_t)INNER * D_ * 2);            // .5 MB
    __bf16* kv_all = (__bf16*)take((size_t)B_ * NC * CHK * KV_N * 2);  // 33 MB
    __bf16* obf    = (__bf16*)take((size_t)B_ * TOPK_ * QL * INNER * 2);
    float*  oproj  = (float*) take((size_t)B_ * TOPK_ * QL * D_ * 4);

    const dim3 blk(256);
    // 1) chunk summaries
    summ_kernel<<<(B_ * NC * D_) / 256, blk, 0, stream>>>(memories, summ);
    // 2) projections (scalar f32; tiny, keeps top-k selection exact)
    linproj_kernel<<<(B_ * QL * D_) / 256, blk, 0, stream>>>(queries, Wsq, bsq, sq, D_, D_, 1.0f);
    linproj_kernel<<<(B_ * NC * D_) / 256, blk, 0, stream>>>(summ, Wsk, bsk, sk, D_, D_, 1.0f);
    linproj_kernel<<<(B_ * QL * INNER) / 256, blk, 0, stream>>>(queries, Wq, nullptr, qproj,
                                                                INNER, D_, 0.125f); // dh^-0.5
    // 3) chunk selection
    topk_kernel<<<B_ * QL, blk, 0, stream>>>(sq, sk, tidx, tw);
    // 4) bf16 operand prep
    prep_abf_kernel<<<(B_ * MLEN * D_) / 256, blk, 0, stream>>>(memories, abf);
    transposeW_kernel<<<(D_ * KV_N) / 256, blk, 0, stream>>>(Wkv, wkvT, D_, KV_N);
    transposeW_kernel<<<(INNER * D_) / 256, blk, 0, stream>>>(Wo, woT, INNER, D_);
    // 5) deduped KV projection for ALL chunks: [16384, 512] x [512, 1024] -> bf16
    {
        dim3 grid(KV_N / 128, (B_ * MLEN) / 64);
        wmma_gemm_kernel<__bf16><<<grid, blk, 0, stream>>>(abf, wkvT, kv_all,
                                                           B_ * MLEN, KV_N, D_);
    }
    // 6) inner attention (wave32 shuffle softmax, one wave per head)
    attn_kernel<<<B_ * QL * TOPK_, blk, 0, stream>>>(qproj, kv_all, tidx, obf);
    // 7) output projection: [2048, 512] x [512, 512] -> f32
    {
        dim3 grid(D_ / 128, (B_ * TOPK_ * QL) / 64);
        wmma_gemm_kernel<float><<<grid, blk, 0, stream>>>(obf, woT, oproj,
                                                          B_ * TOPK_ * QL, D_, D_);
    }
    // 8) top-k weighted combine + bias
    combine_kernel<<<(B_ * QL * D_) / 256, blk, 0, stream>>>(oproj, tw, bo, out);
}
